// AttentionHiddenFusion_37048387895870
// MI455X (gfx1250) — compile-verified, matrix operand order checked by reading, not run on hardware
//
#include <hip/hip_runtime.h>
#include <math.h>

typedef __attribute__((ext_vector_type(16))) _Float16 v16h;
typedef __attribute__((ext_vector_type(8)))  float    v8f;
typedef __attribute__((ext_vector_type(4)))  float    f4;

#define B_ 16
#define S_ 4096
#define H_ 1024
#define A_ 1024
#define D_ 6
#define LN_EPS 1e-5f
#define CLS_SCALE 1.0f

// workspace layout (floats)
#define WCOMB_OFF  0        // 7 x 1024 : gamma-folded Wr rows (d=0..5) + Wtg (d=6)
#define SG_OFF     7168     // 7 : sum(gamma*w)
#define SBETA_OFF  7175     // 7 : dot(beta, w)
#define CLSCTX_OFF 7182     // 16*6 : CLS context per (b,d)
#define GATE_OFF   7278     // 16 : per-batch gate scalar

#define LOG2E_X2 2.88539008177792681f   // 2*log2(e)
#define LOG2E    1.44269504088896340f

__device__ __forceinline__ float fast_tanh(float x) {
#if __has_builtin(__builtin_amdgcn_tanhf)
    return __builtin_amdgcn_tanhf(x);
#else
    // tanh(x) = 1 - 2/(e^{2x}+1); branch-free, correct at +-inf via rcp/exp2 semantics
    float e = __builtin_amdgcn_exp2f(x * LOG2E_X2);
    return 1.0f - 2.0f * __builtin_amdgcn_rcpf(e + 1.0f);
#endif
}
__device__ __forceinline__ float fast_sigmoid(float x) {
    // 1/(1+e^{-x}) branch-free
    float e = __builtin_amdgcn_exp2f(-x * LOG2E);
    return __builtin_amdgcn_rcpf(1.0f + e);
}
__device__ __forceinline__ float gelu_exact(float x) {
    return 0.5f * x * (1.0f + erff(x * 0.70710678118654752f));
}

// ---------------- setup kernel 1: fold gamma into projection weights --------
__global__ __launch_bounds__(256)
void prep_weights(const float* __restrict__ gamma, const float* __restrict__ beta,
                  const float* __restrict__ Wr, const float* __restrict__ Wtg,
                  float* __restrict__ ws) {
    int d = blockIdx.x;                       // 0..6
    const float* w = (d < 6) ? (Wr + d * A_) : Wtg;
    float* wc = ws + WCOMB_OFF + d * A_;
    __shared__ float sg, sb;
    if (threadIdx.x == 0) { sg = 0.f; sb = 0.f; }
    __syncthreads();
    float psg = 0.f, psb = 0.f;
    for (int i = threadIdx.x; i < A_; i += blockDim.x) {
        float g = gamma[i] * w[i];
        wc[i] = g;
        psg += g;
        psb += beta[i] * w[i];
    }
    atomicAdd(&sg, psg);
    atomicAdd(&sb, psb);
    __syncthreads();
    if (threadIdx.x == 0) { ws[SG_OFF + d] = sg; ws[SBETA_OFF + d] = sb; }
}

// ---------------- setup kernel 2: per-batch CLS context + scalar gate -------
__global__ __launch_bounds__(256)
void prep_cls(const float* __restrict__ attn, const float* __restrict__ gamma,
              const float* __restrict__ beta, const float* __restrict__ Wr,
              const float* __restrict__ Wc, const float* __restrict__ Wsg,
              const float* __restrict__ bsg, float* __restrict__ ws) {
    int b = blockIdx.x;
    const float* x = attn + (size_t)b * S_ * A_;   // row s=0
    __shared__ float red0, red1;
    __shared__ float acc[12];
    int tid = threadIdx.x;
    if (tid == 0) { red0 = 0.f; red1 = 0.f; }
    if (tid < 12) acc[tid] = 0.f;
    __syncthreads();
    float xv[4];
    float s = 0.f, q = 0.f;
    #pragma unroll
    for (int c = 0; c < 4; c++) {
        xv[c] = x[tid * 4 + c];
        s += xv[c]; q += xv[c] * xv[c];
    }
    atomicAdd(&red0, s);
    atomicAdd(&red1, q);
    __syncthreads();
    float mu   = red0 * (1.0f / A_);
    float var  = red1 * (1.0f / A_) - mu * mu;
    float rstd = rsqrtf(var + LN_EPS);
    float part[12];
    #pragma unroll
    for (int j = 0; j < 12; j++) part[j] = 0.f;
    #pragma unroll
    for (int c = 0; c < 4; c++) {
        int i = tid * 4 + c;
        float n = (xv[c] - mu) * rstd * gamma[i] + beta[i];
        #pragma unroll
        for (int j = 0; j < 6; j++) {
            part[j]     += n * Wr[j * A_ + i];
            part[6 + j] += n * Wc[j * A_ + i];
        }
    }
    #pragma unroll
    for (int j = 0; j < 12; j++) atomicAdd(&acc[j], part[j]);
    __syncthreads();
    if (tid == 0) {
        float gsum = 0.f;
        #pragma unroll
        for (int d = 0; d < 6; d++) {
            float ctx = CLS_SCALE * acc[6 + d];
            ws[CLSCTX_OFF + b * 6 + d] = ctx;
            float th = acc[d] + ctx;           // token_hidden[b,0,d] pre-gelu
            gsum += gelu_exact(th) * Wsg[d];
        }
        ws[GATE_OFF + b] = fast_sigmoid(gsum + bsg[0]);
    }
}

// ---------------- main fused kernel: 16 s-rows per workgroup ----------------
__global__ __launch_bounds__(256)
void fused_main(const float* __restrict__ hidden, const float* __restrict__ attn,
                const float* __restrict__ We, const float* __restrict__ btg,
                const float* __restrict__ layer_scale, const float* __restrict__ ws,
                float* __restrict__ out) {
    int blk  = blockIdx.x;
    int b    = blk >> 8;            // S/16 = 256 tiles per batch
    int tile = blk & 255;
    int s0   = tile * 16;
    int wave = threadIdx.x >> 5;
    int lane = threadIdx.x & 31;

    __shared__ _Float16 thA[16][8];     // A-matrix rows (K=0..5 live, 6..7 zero)
    __shared__ float    rowCoef[16];    // gate[b] * token_gate[row]

    const float gate_b = ws[GATE_OFF + b];
    const float btg0   = btg[0];

    // ---- stage 1: LayerNorm + 7 projections, 2 rows per wave, single pass
    for (int rr = 0; rr < 2; rr++) {
        int row = wave * 2 + rr;
        const float* x = attn + ((size_t)(b * S_ + s0 + row)) * A_;
        float s = 0.f, q = 0.f;
        float s1[7];
        #pragma unroll
        for (int d = 0; d < 7; d++) s1[d] = 0.f;
        #pragma unroll
        for (int k = 0; k < 8; k++) {
            int i4 = k * 32 + lane;                       // float4 index, coalesced
            f4 xv = __builtin_nontemporal_load(((const f4*)x) + i4);  // streaming
            s += xv.x + xv.y + xv.z + xv.w;
            q += xv.x * xv.x + xv.y * xv.y + xv.z * xv.z + xv.w * xv.w;
            #pragma unroll
            for (int d = 0; d < 7; d++) {
                f4 wv = ((const f4*)(ws + WCOMB_OFF + d * A_))[i4];   // cached
                s1[d] += xv.x * wv.x + xv.y * wv.y + xv.z * wv.z + xv.w * wv.w;
            }
        }
        // wave32 butterfly reductions
        #pragma unroll
        for (int off = 16; off > 0; off >>= 1) {
            s += __shfl_xor(s, off, 32);
            q += __shfl_xor(q, off, 32);
            #pragma unroll
            for (int d = 0; d < 7; d++) s1[d] += __shfl_xor(s1[d], off, 32);
        }
        if (lane == 0) {
            float mu   = s * (1.0f / A_);
            float var  = q * (1.0f / A_) - mu * mu;
            float rstd = rsqrtf(var + LN_EPS);
            #pragma unroll
            for (int d = 0; d < 6; d++) {
                float th = rstd * (s1[d] - mu * ws[SG_OFF + d]) + ws[SBETA_OFF + d]
                         + ws[CLSCTX_OFF + b * 6 + d];
                thA[row][d] = (_Float16)gelu_exact(th);
            }
            thA[row][6] = (_Float16)0.f;
            thA[row][7] = (_Float16)0.f;
            float tgp = rstd * (s1[6] - mu * ws[SG_OFF + 6]) + ws[SBETA_OFF + 6] + btg0;
            rowCoef[row] = gate_b * fast_sigmoid(tgp);
        }
    }
    __syncthreads();

    // ---- build WMMA A fragment (16x32 f16; only K=0..5, lanes 0..15 live)
    int  lc = lane & 15;
    bool lo = lane < 16;
    v16h afrag = {};
    #pragma unroll
    for (int j = 0; j < 6; j++) {
        _Float16 v = thA[lc][j];               // safe address for all lanes
        afrag[j] = lo ? v : (_Float16)0.f;
    }

    // ---- stage 2: 8 N-tiles per wave -> 64 tiles cover H=1024
    for (int t = 0; t < 8; t++) {
        int h0 = (wave + t * 8) * 16;
        int h  = h0 + lc;
        // B fragment: 32x16 f16, B[k][n] = We[h][k]; lanes 0..15, elements 0..5 = K 0..5
        v16h bfrag = {};
        const float* wp = We + h * D_;
        #pragma unroll
        for (int j = 0; j < 6; j++) {
            float wv = wp[j];
            bfrag[j] = lo ? (_Float16)wv : (_Float16)0.f;
        }
        v8f c = {};
        c = __builtin_amdgcn_wmma_f32_16x16x32_f16(
                /*neg_a=*/false, afrag, /*neg_b=*/false, bfrag,
                /*c_mod=*/(short)0, c, /*reuse_a=*/false, /*reuse_b=*/false);

        float ls    = layer_scale[h];
        int   mbase = lo ? 0 : 8;              // C layout: lanes>=16 hold M=r+8
        #pragma unroll
        for (int r = 0; r < 8; r++) {
            int M = mbase + r;
            size_t idx = ((size_t)(b * S_ + s0 + M)) * H_ + h;
            float mod = fast_tanh(c[r]);                       // branch-free
            float hid = __builtin_nontemporal_load(hidden + idx);
            float o   = hid * (1.0f + rowCoef[M] * ls * mod);
            __builtin_nontemporal_store(o, out + idx);
        }
    }
}

extern "C" void kernel_launch(void* const* d_in, const int* in_sizes, int n_in,
                              void* d_out, int out_size, void* d_ws, size_t ws_size,
                              hipStream_t stream) {
    const float* hidden      = (const float*)d_in[0];
    const float* attn_out    = (const float*)d_in[1];
    const float* ln_gamma    = (const float*)d_in[2];
    const float* ln_beta     = (const float*)d_in[3];
    const float* Wr          = (const float*)d_in[4];
    const float* Wc          = (const float*)d_in[5];
    const float* We          = (const float*)d_in[6];
    const float* Wtg         = (const float*)d_in[7];
    const float* btg         = (const float*)d_in[8];
    const float* Wsg         = (const float*)d_in[9];
    const float* bsg         = (const float*)d_in[10];
    const float* layer_scale = (const float*)d_in[11];
    float* out = (float*)d_out;
    float* ws  = (float*)d_ws;

    hipLaunchKernelGGL(prep_weights, dim3(7),  dim3(256), 0, stream,
                       ln_gamma, ln_beta, Wr, Wtg, ws);
    hipLaunchKernelGGL(prep_cls,    dim3(16), dim3(256), 0, stream,
                       attn_out, ln_gamma, ln_beta, Wr, Wc, Wsg, bsg, ws);
    hipLaunchKernelGGL(fused_main,  dim3((B_ * S_) / 16), dim3(256), 0, stream,
                       hidden, attn_out, We, btg, layer_scale, ws, out);
}
